// SelfAttention_73693048865292
// MI455X (gfx1250) — matrix operand order repaired and checked by reading, not compiled, not run on hardware
//
#include <hip/hip_runtime.h>
#include <hip/hip_bf16.h>
#include <stdint.h>

// ---------------------------------------------------------------------------
// Problem constants (from reference): B=8, C=256, H=W=32 -> S=1024,
// HEADS=4, HD=64, GROUPS=32 (8 channels/group), EPS=1e-5.
// ---------------------------------------------------------------------------
#define NB     8
#define NC     256
#define NS     1024
#define NHEADS 4
#define HD     64
#define NGROUP 32

typedef __bf16 v16bf __attribute__((ext_vector_type(16)));
typedef float  v8f   __attribute__((ext_vector_type(8)));

union BFQ { uint4 q[2]; v16bf v; };

__device__ __forceinline__ v8f wmma_bf16(const BFQ& a, const BFQ& b, v8f c) {
    // D = A(16x32 bf16) * B(32x16 bf16) + C(16x16 f32)
    return __builtin_amdgcn_wmma_f32_16x16x32_bf16(
        false, a.v, false, b.v, (short)0, c, false, false);
}

__device__ __forceinline__ unsigned short f2bf(float f) {
    unsigned u = __builtin_bit_cast(unsigned, f);
    u += 0x7fffu + ((u >> 16) & 1u);   // round-to-nearest-even
    return (unsigned short)(u >> 16);
}

// ---- gfx1250 async global->LDS path (guarded; falls back to sync copy) ----
#if defined(__has_builtin)
#if __has_builtin(__builtin_amdgcn_global_load_async_to_lds_b128) && \
    __has_builtin(__builtin_amdgcn_s_wait_asynccnt)
#define HAVE_ASYNC_LDS 1
#endif
#endif

#if defined(HAVE_ASYNC_LDS)
// Builtin signature (from hipcc diagnostic): param 0 is
// 'int __attribute__((vector_size(16))) __device__ *'  (AS(1) global).
typedef int b128v __attribute__((vector_size(16)));
typedef b128v __attribute__((address_space(1)))* async_gptr;
typedef b128v __attribute__((address_space(3)))* async_lptr;
#endif

__device__ __forceinline__ void stage_b128(const unsigned short* g,
                                           unsigned short* l) {
#if defined(HAVE_ASYNC_LDS)
    __builtin_amdgcn_global_load_async_to_lds_b128(
        (async_gptr)g, (async_lptr)l, 0, 0);
#else
    *(uint4*)l = *(const uint4*)g;
#endif
}

__device__ __forceinline__ void stage_wait() {
#if defined(HAVE_ASYNC_LDS)
    __builtin_amdgcn_s_wait_asynccnt(0);
#endif
}

// ---------------------------------------------------------------------------
// Kernel 0: convert w_qkv (768x256) and w_proj (256x256) f32 -> bf16
// ---------------------------------------------------------------------------
__global__ void k_convert_w(const float* wqkv, const float* wproj,
                            unsigned short* wqkv_bf, unsigned short* wproj_bf) {
    int i = blockIdx.x * blockDim.x + threadIdx.x;
    const int NQ = 768 * 256;
    if (i < NQ) {
        wqkv_bf[i] = f2bf(wqkv[i]);
    } else {
        int j = i - NQ;               // j < 256*256
        wproj_bf[j] = f2bf(wproj[j]);
    }
}

// ---------------------------------------------------------------------------
// Kernel 1: GroupNorm.  One block per (b, group).  Output h_t is bf16 in
// [b][s][c] layout (transposed) so GEMM1's B-operand loads are b128s.
// ---------------------------------------------------------------------------
__global__ void k_groupnorm(const float* __restrict__ x,
                            const float* __restrict__ gamma,
                            const float* __restrict__ beta,
                            unsigned short* __restrict__ h_t) {
    int bg = blockIdx.x;
    int b = bg >> 5, g = bg & 31;
    const float* xp = x + ((size_t)(b * NC + g * 8)) * NS;   // 8 x 1024 slab

    float s = 0.f, ss = 0.f;
    for (int i = threadIdx.x; i < 8 * NS; i += 256) {
        float v = xp[i];
        s += v; ss += v * v;
    }
    #pragma unroll
    for (int m = 16; m > 0; m >>= 1) {
        s  += __shfl_xor(s,  m, 32);
        ss += __shfl_xor(ss, m, 32);
    }
    __shared__ float red[2][8];
    int wv = threadIdx.x >> 5, ln = threadIdx.x & 31;
    if (ln == 0) { red[0][wv] = s; red[1][wv] = ss; }
    __syncthreads();
    s  = red[0][ln & 7];
    ss = red[1][ln & 7];
    #pragma unroll
    for (int m = 4; m > 0; m >>= 1) {
        s  += __shfl_xor(s,  m, 32);
        ss += __shfl_xor(ss, m, 32);
    }
    const float inv_n = 1.0f / (8.0f * NS);
    float mean = s * inv_n;
    float var  = ss * inv_n - mean * mean;
    float rstd = rsqrtf(var + 1e-5f);

    for (int i = threadIdx.x; i < 8 * NS; i += 256) {
        int cl = i >> 10;            // 0..7
        int sp = i & 1023;
        int c  = g * 8 + cl;
        float v = (xp[i] - mean) * rstd * gamma[c] + beta[c];
        h_t[((size_t)b * NS + sp) * NC + c] = f2bf(v);
    }
}

// ---------------------------------------------------------------------------
// Kernel 2: QKV GEMM.  qkv[o,s] = sum_c Wqkv[o,c] * h[c,s]  (per batch).
// One wave computes a 16(o) x 64(s) tile.  Epilogue scatters into
// q_t [bh][s][64] (x0.125), k_t [bh][s][64], v_c [bh][64][s].
// ---------------------------------------------------------------------------
__global__ void k_gemm_qkv(const unsigned short* __restrict__ wq,
                           const float* __restrict__ bq,
                           const unsigned short* __restrict__ h_t,
                           unsigned short* __restrict__ q_t,
                           unsigned short* __restrict__ k_t,
                           unsigned short* __restrict__ v_c) {
    int w    = (blockIdx.x << 2) | (threadIdx.x >> 5);   // global wave id
    int lane = threadIdx.x & 31;
    int b   = w / 768;            // 48 o-tiles * 16 s-chunks
    int rem = w % 768;
    int o0  = (rem >> 4) * 16;
    int s0  = (rem & 15) * 64;
    int lr   = lane & 15;
    int hsel = (lane >> 4) & 1;

    v8f acc[4] = {};
    const unsigned short* hb = h_t + (size_t)b * NS * NC;

    for (int kk = 0; kk < 8; ++kk) {
        int k0 = kk * 32;
        BFQ A;
        const unsigned short* ap = wq + (size_t)(o0 + lr) * NC + k0 + hsel * 8;
        A.q[0] = *(const uint4*)ap;
        A.q[1] = *(const uint4*)(ap + 16);
        #pragma unroll
        for (int n = 0; n < 4; ++n) {
            int s = s0 + n * 16 + lr;
            const unsigned short* bp = hb + (size_t)s * NC + k0 + hsel * 8;
            BFQ Bm;
            Bm.q[0] = *(const uint4*)bp;
            Bm.q[1] = *(const uint4*)(bp + 16);
            acc[n] = wmma_bf16(A, Bm, acc[n]);
        }
    }

    int rbase = hsel * 8;
    #pragma unroll
    for (int n = 0; n < 4; ++n) {
        int s = s0 + n * 16 + lr;
        #pragma unroll
        for (int v = 0; v < 8; ++v) {
            int o = o0 + rbase + v;              // 0..767
            float val = acc[n][v] + bq[o];
            int which = o >> 8;                  // 0=q 1=k 2=v
            int wi    = o & 255;
            int head  = wi >> 6;
            int cl    = wi & 63;
            int bh    = b * NHEADS + head;
            if (which == 0)
                q_t[((size_t)bh * NS + s) * HD + cl] = f2bf(val * 0.125f);
            else if (which == 1)
                k_t[((size_t)bh * NS + s) * HD + cl] = f2bf(val);
            else
                v_c[((size_t)bh * HD + cl) * NS + s] = f2bf(val);
        }
    }
}

// ---------------------------------------------------------------------------
// Kernel 3: flash attention with block-shared K/V staged through LDS.
// Block = 4 waves = 4 consecutive 16-row query tiles of ONE (b,head).
// Per 32-key chunk: stage K (32x64 bf16) + V (64x32 bf16) into a double
// buffer (async global->LDS when available), then each wave does
// 4 WMMAs for S = q^T k, online softmax, P transpose via per-wave LDS slab,
// 4 WMMAs for O += P V, reading all B-operands with ds_load_b128.
// ---------------------------------------------------------------------------
__global__ void k_attention(const unsigned short* __restrict__ q_t,
                            const unsigned short* __restrict__ k_t,
                            const unsigned short* __restrict__ v_c,
                            unsigned short* __restrict__ ao_t) {
    __shared__ __align__(16) unsigned short kbuf[2][32][HD];   // 8 KB
    __shared__ __align__(16) unsigned short vbuf[2][HD][32];   // 8 KB
    __shared__ __align__(16) unsigned short pbuf[4][16 * 32];  // 4 KB

    int tid  = threadIdx.x;
    int lane = tid & 31;
    int wl   = tid >> 5;
    int bh   = blockIdx.x >> 4;                 // 0..31  (b*4+head)
    int i0   = ((blockIdx.x & 15) * 4 + wl) * 16;
    int lr   = lane & 15;
    int hsel = (lane >> 4) & 1;

    const unsigned short* qb = q_t + (size_t)bh * NS * HD;
    const unsigned short* kb = k_t + (size_t)bh * NS * HD;
    const unsigned short* vb = v_c + (size_t)bh * HD * NS;

    // staging coordinates for this thread (K: 4 threads/row, V: 2 threads/row)
    int kr = tid >> 2, kp16 = (tid & 3) * 16;          // kbuf[.][kr][kp16..+15]
    int vr = tid >> 1, vp16 = (tid & 1) * 16;          // vbuf[.][vr][vp16..+15]

    // q in A-layout for both K halves (c 0..31 and 32..63); scale pre-folded
    BFQ qa0, qa1;
    {
        const unsigned short* qp = qb + (size_t)(i0 + lr) * HD + hsel * 8;
        qa0.q[0] = *(const uint4*)(qp);
        qa0.q[1] = *(const uint4*)(qp + 16);
        qa1.q[0] = *(const uint4*)(qp + 32);
        qa1.q[1] = *(const uint4*)(qp + 48);
    }

    v8f O[4] = {};
    float m[8], l[8];
    #pragma unroll
    for (int v = 0; v < 8; ++v) { m[v] = -1e30f; l[v] = 0.f; }

    unsigned short* pb = pbuf[wl];

    // ---- prologue: stage chunk 0 into buffer 0 ----
    {
        const unsigned short* kg = kb + (size_t)kr * HD + kp16;
        stage_b128(kg,     &kbuf[0][kr][kp16]);
        stage_b128(kg + 8, &kbuf[0][kr][kp16 + 8]);
        const unsigned short* vg = vb + (size_t)vr * NS + vp16;
        stage_b128(vg,     &vbuf[0][vr][vp16]);
        stage_b128(vg + 8, &vbuf[0][vr][vp16 + 8]);
    }

    for (int it = 0; it < 32; ++it) {
        int buf = it & 1;
        int j0  = it * 32;

        stage_wait();          // our async loads for buf are complete
        __syncthreads();       // everyone's loads done; prev-buf reads done

        if (it + 1 < 32) {
            int jn = j0 + 32;
            const unsigned short* kg = kb + (size_t)(jn + kr) * HD + kp16;
            stage_b128(kg,     &kbuf[buf ^ 1][kr][kp16]);
            stage_b128(kg + 8, &kbuf[buf ^ 1][kr][kp16 + 8]);
            const unsigned short* vg = vb + (size_t)vr * NS + jn + vp16;
            stage_b128(vg,     &vbuf[buf ^ 1][vr][vp16]);
            stage_b128(vg + 8, &vbuf[buf ^ 1][vr][vp16 + 8]);
        }

        // ---- S = q^T k for 32 keys (two 16x16 D tiles), B from LDS ----
        v8f S0 = {}, S1 = {};
        {
            BFQ kB;
            const unsigned short* kp0 = &kbuf[buf][lr][hsel * 8];
            kB.q[0] = *(const uint4*)kp0;
            kB.q[1] = *(const uint4*)(kp0 + 16);
            S0 = wmma_bf16(qa0, kB, S0);
            kB.q[0] = *(const uint4*)(kp0 + 32);
            kB.q[1] = *(const uint4*)(kp0 + 48);
            S0 = wmma_bf16(qa1, kB, S0);
            const unsigned short* kp1 = &kbuf[buf][16 + lr][hsel * 8];
            kB.q[0] = *(const uint4*)kp1;
            kB.q[1] = *(const uint4*)(kp1 + 16);
            S1 = wmma_bf16(qa0, kB, S1);
            kB.q[0] = *(const uint4*)(kp1 + 32);
            kB.q[1] = *(const uint4*)(kp1 + 48);
            S1 = wmma_bf16(qa1, kB, S1);
        }

        // ---- online softmax per row (row = hsel*8 + v) ----
        float P0[8], P1[8];
        #pragma unroll
        for (int v = 0; v < 8; ++v) {
            float t = fmaxf(S0[v], S1[v]);
            t = fmaxf(t, __shfl_xor(t, 1, 32));
            t = fmaxf(t, __shfl_xor(t, 2, 32));
            t = fmaxf(t, __shfl_xor(t, 4, 32));
            t = fmaxf(t, __shfl_xor(t, 8, 32));
            float mn    = fmaxf(m[v], t);
            float alpha = __expf(m[v] - mn);
            float p0 = __expf(S0[v] - mn);
            float p1 = __expf(S1[v] - mn);
            float sum = p0 + p1;
            sum += __shfl_xor(sum, 1, 32);
            sum += __shfl_xor(sum, 2, 32);
            sum += __shfl_xor(sum, 4, 32);
            sum += __shfl_xor(sum, 8, 32);
            l[v] = l[v] * alpha + sum;
            m[v] = mn;
            O[0][v] *= alpha; O[1][v] *= alpha; O[2][v] *= alpha; O[3][v] *= alpha;
            P0[v] = p0; P1[v] = p1;
        }

        // ---- transpose P (C/D layout -> A layout) via per-wave LDS slab ----
        #pragma unroll
        for (int v = 0; v < 8; ++v) {
            int r = (hsel << 3) + v;
            pb[r * 32 + lr]      = f2bf(P0[v]);
            pb[r * 32 + lr + 16] = f2bf(P1[v]);
        }
        BFQ P;
        P.q[0] = *(const uint4*)(pb + lr * 32 + hsel * 8);
        P.q[1] = *(const uint4*)(pb + lr * 32 + hsel * 8 + 16);

        // ---- O += P * V (four c-tiles of 16), B from LDS ----
        #pragma unroll
        for (int ct = 0; ct < 4; ++ct) {
            const unsigned short* vp = &vbuf[buf][ct * 16 + lr][hsel * 8];
            BFQ Vb;
            Vb.q[0] = *(const uint4*)vp;
            Vb.q[1] = *(const uint4*)(vp + 16);
            O[ct] = wmma_bf16(P, Vb, O[ct]);
        }
    }

    // ---- finalize: divide by row sums, store ao_t [b][s][256] bf16 ----
    int b = bh >> 2, head = bh & 3;
    #pragma unroll
    for (int ct = 0; ct < 4; ++ct) {
        int c = head * HD + ct * 16 + lr;
        #pragma unroll
        for (int v = 0; v < 8; ++v) {
            int i = i0 + (hsel << 3) + v;
            ao_t[((size_t)b * NS + i) * NC + c] = f2bf(O[ct][v] / l[v]);
        }
    }
}

// ---------------------------------------------------------------------------
// Kernel 4: output projection + residual.  out = x + Wproj * ao + b_proj.
// ---------------------------------------------------------------------------
__global__ void k_gemm_proj(const unsigned short* __restrict__ wp,
                            const float* __restrict__ bp,
                            const unsigned short* __restrict__ ao_t,
                            const float* __restrict__ x,
                            float* __restrict__ out) {
    int w    = (blockIdx.x << 2) | (threadIdx.x >> 5);
    int lane = threadIdx.x & 31;
    int b   = w >> 8;                 // 16 o-tiles * 16 s-chunks = 256/batch
    int rem = w & 255;
    int o0  = (rem >> 4) * 16;
    int s0  = (rem & 15) * 64;
    int lr   = lane & 15;
    int hsel = (lane >> 4) & 1;

    v8f acc[4] = {};
    const unsigned short* ab = ao_t + (size_t)b * NS * NC;

    for (int kk = 0; kk < 8; ++kk) {
        int k0 = kk * 32;
        BFQ A;
        const unsigned short* ap = wp + (size_t)(o0 + lr) * NC + k0 + hsel * 8;
        A.q[0] = *(const uint4*)ap;
        A.q[1] = *(const uint4*)(ap + 16);
        #pragma unroll
        for (int n = 0; n < 4; ++n) {
            int s = s0 + n * 16 + lr;
            const unsigned short* bpr = ab + (size_t)s * NC + k0 + hsel * 8;
            BFQ Bm;
            Bm.q[0] = *(const uint4*)bpr;
            Bm.q[1] = *(const uint4*)(bpr + 16);
            acc[n] = wmma_bf16(A, Bm, acc[n]);
        }
    }

    int rbase = hsel * 8;
    #pragma unroll
    for (int n = 0; n < 4; ++n) {
        int s = s0 + n * 16 + lr;
        #pragma unroll
        for (int v = 0; v < 8; ++v) {
            int o = o0 + rbase + v;
            size_t idx = ((size_t)(b * NC + o)) * NS + s;
            out[idx] = x[idx] + acc[n][v] + bp[o];
        }
    }
}

// ---------------------------------------------------------------------------
// Host-side launcher
// ---------------------------------------------------------------------------
extern "C" void kernel_launch(void* const* d_in, const int* in_sizes, int n_in,
                              void* d_out, int out_size, void* d_ws, size_t ws_size,
                              hipStream_t stream) {
    const float* x      = (const float*)d_in[0];
    const float* gamma  = (const float*)d_in[1];
    const float* beta   = (const float*)d_in[2];
    const float* w_qkv  = (const float*)d_in[3];
    const float* b_qkv  = (const float*)d_in[4];
    const float* w_proj = (const float*)d_in[5];
    const float* b_proj = (const float*)d_in[6];
    float* out = (float*)d_out;

    // workspace layout (bytes), ~20.5 MB total
    char* ws = (char*)d_ws;
    unsigned short* wqkv_bf  = (unsigned short*)(ws + 0);           // 768*256*2
    unsigned short* wproj_bf = (unsigned short*)(ws + 393216);      // 256*256*2
    unsigned short* h_t      = (unsigned short*)(ws + 524288);      // 8*1024*256*2
    unsigned short* q_t      = (unsigned short*)(ws + 4718592);     // 8*4*1024*64*2
    unsigned short* k_t      = (unsigned short*)(ws + 8912896);
    unsigned short* v_c      = (unsigned short*)(ws + 13107200);
    unsigned short* ao_t     = (unsigned short*)(ws + 17301504);

    // 0: weight conversion  (768*256 + 256*256 = 262144 elements)
    k_convert_w<<<1024, 256, 0, stream>>>(w_qkv, w_proj, wqkv_bf, wproj_bf);

    // 1: group norm -> h_t   (one block per (b, group))
    k_groupnorm<<<NB * NGROUP, 256, 0, stream>>>(x, gamma, beta, h_t);

    // 2: QKV GEMM   (8 batches * 48 o-tiles * 16 s-chunks = 6144 waves)
    k_gemm_qkv<<<1536, 128, 0, stream>>>(wqkv_bf, b_qkv, h_t, q_t, k_t, v_c);

    // 3: flash attention (32 bh * 16 blocks, 4 i-tiles/block share K/V via LDS)
    k_attention<<<512, 128, 0, stream>>>(q_t, k_t, v_c, ao_t);

    // 4: projection + residual (8 * 16 * 16 = 2048 waves)
    k_gemm_proj<<<512, 128, 0, stream>>>(wproj_bf, b_proj, ao_t, x, out);
}
